// MHSA_80324478369904
// MI455X (gfx1250) — matrix operand order repaired and checked by reading, not compiled, not run on hardware
//
#include <hip/hip_runtime.h>
#include <hip/hip_bf16.h>
#include <math.h>

// ---- problem constants (match reference) ----
#define BB   8
#define SS   1024
#define DD   1152
#define HH   16
#define DH_  72
#define DHP  96    // q/k head-dim padded to 3 K-chunks of 32
#define DHV  80    // v head-dim padded to 5 N-tiles of 16
#define EPSN 1e-5f

typedef __attribute__((ext_vector_type(16))) _Float16 vh16;
typedef __attribute__((ext_vector_type(8)))  _Float16 vh8;
typedef __attribute__((ext_vector_type(8)))  float    vf8;
typedef __attribute__((ext_vector_type(4)))  unsigned int vu4;
typedef __attribute__((ext_vector_type(8)))  int vi8;
typedef __attribute__((ext_vector_type(4)))  int vi4;

#if __has_builtin(__builtin_amdgcn_tensor_load_to_lds)
#define HAVE_TDM 1
#else
#define HAVE_TDM 0
#endif

__device__ __forceinline__ vf8 wmma16(vh16 a, vh16 b, vf8 c) {
  // D = A(16x32 f16) * B(32x16 f16) + C(16x16 f32)
  return __builtin_amdgcn_wmma_f32_16x16x32_f16(false, a, false, b, (short)0, c,
                                                false, false);
}

// A operand k order used here (B loads match): lane row m holds K = {kb..kb+15},
// kb = 0 for lanes 0-15, 16 for lanes 16-31.  Two 16B loads, pack to v16h.
__device__ __forceinline__ vh16 packA(const _Float16* p) {
  vh8 lo = *(const vh8*)p;
  vh8 hi = *(const vh8*)(p + 8);
  vh16 a;
#pragma unroll
  for (int e = 0; e < 8; ++e) { a[e] = lo[e]; a[e + 8] = hi[e]; }
  return a;
}

#if HAVE_TDM
// Issue one TDM 2D tile load: global (row-major, 2-byte elems) -> LDS.
// D# per CDNA5 ISA. Tracked by TENSORcnt.
__device__ __forceinline__ void tdm_load_2d(unsigned lds_byte_off,
                                            const void* gaddr,
                                            unsigned rowElems,     // tile_dim0
                                            unsigned rows,         // tile_dim1
                                            unsigned strideElems) {// dim0 stride
  unsigned long long ga = (unsigned long long)(size_t)gaddr;
  vu4 g0;
  g0[0] = 1u;                                   // count=1, user descriptor
  g0[1] = lds_byte_off;                         // lds_addr
  g0[2] = (unsigned)ga;                         // global_addr[31:0]
  g0[3] = (unsigned)(ga >> 32) | (2u << 30);    // global_addr[56:32] | type=2
  vi8 g1;
  g1[0] = (int)(1u << 16);                      // data_size=1 -> 2 bytes
  g1[1] = (int)((rowElems & 0xFFFFu) << 16);    // tensor_dim0[15:0]
  g1[2] = (int)((rowElems >> 16) | ((rows & 0xFFFFu) << 16)); // td0[31:16]|td1[15:0]
  g1[3] = (int)((rows >> 16) | ((rowElems & 0xFFFFu) << 16)); // td1[31:16]|tile_dim0
  g1[4] = (int)(rows & 0xFFFFu);                // tile_dim1 (tile_dim2 = 0)
  g1[5] = (int)strideElems;                     // tensor_dim0_stride[31:0]
  g1[6] = 0;
  g1[7] = 0;
  vi4 z4 = {};
#if defined(__clang_major__) && (__clang_major__ >= 23)
  vi8 z8 = {};
  __builtin_amdgcn_tensor_load_to_lds(g0, g1, z4, z4, z8, 0);
#else
  __builtin_amdgcn_tensor_load_to_lds(g0, g1, z4, z4, 0);
#endif
}

__device__ __forceinline__ void tdm_wait0() {
#if __has_builtin(__builtin_amdgcn_s_wait_tensorcnt)
  __builtin_amdgcn_s_wait_tensorcnt(0);
#else
  asm volatile("s_wait_tensorcnt 0" ::: "memory");
#endif
}
#endif

// ---------------------------------------------------------------------------
// Kernel 1: non-affine LayerNorm + AdaLN modulation, f32 -> f16 h
// ---------------------------------------------------------------------------
__global__ void __launch_bounds__(256, 1)
adaln_kernel(const float* __restrict__ x,
             const float* __restrict__ gamma,
             const float* __restrict__ beta,
             _Float16* __restrict__ h) {
  const int row = blockIdx.x;               // 0 .. B*S-1
  const int b   = row / SS;
  const float* xr = x + (size_t)row * DD;
  const int tid = threadIdx.x;

  float vals[5];
  float s = 0.f, ss = 0.f;
#pragma unroll
  for (int i = 0; i < 5; ++i) {
    int idx = tid + i * 256;
    float v = (idx < DD) ? xr[idx] : 0.f;
    vals[i] = v; s += v; ss += v * v;
  }
#pragma unroll
  for (int m = 16; m >= 1; m >>= 1) { s += __shfl_xor(s, m); ss += __shfl_xor(ss, m); }
  __shared__ float rs[8], rss[8];
  const int wid = tid >> 5, lane = tid & 31;
  if (lane == 0) { rs[wid] = s; rss[wid] = ss; }
  __syncthreads();
  if (wid == 0) {
    float a = (lane < 8) ? rs[lane] : 0.f;
    float c = (lane < 8) ? rss[lane] : 0.f;
#pragma unroll
    for (int m = 4; m >= 1; m >>= 1) { a += __shfl_xor(a, m); c += __shfl_xor(c, m); }
    if (lane == 0) { rs[0] = a; rss[0] = c; }
  }
  __syncthreads();
  const float mean = rs[0] * (1.f / DD);
  const float var  = rss[0] * (1.f / DD) - mean * mean;
  const float rstd = rsqrtf(var + EPSN);

#pragma unroll
  for (int i = 0; i < 5; ++i) {
    int idx = tid + i * 256;
    if (idx < DD) {
      float n = (vals[i] - mean) * rstd;
      float g = gamma[(size_t)b * DD + idx];
      float be = beta[(size_t)b * DD + idx];
      h[(size_t)row * DD + idx] = (_Float16)(n * (1.f + g) + be);
    }
  }
}

// ---------------------------------------------------------------------------
// Kernel 2: convert Wq/Wk/Wv/Wo f32 [k][n] -> f16 transposed WT[n][k],
// then one zero-filled pad row (D halves) used by padded head columns.
// ---------------------------------------------------------------------------
__global__ void __launch_bounds__(256, 1)
wtrans_kernel(const float* __restrict__ Wq,
              const float* __restrict__ Wk,
              const float* __restrict__ Wv,
              const float* __restrict__ Wo,
              _Float16* __restrict__ WT) {
  size_t idx = (size_t)blockIdx.x * 256 + threadIdx.x;   // 4*D*D + D total
  const size_t per = (size_t)DD * DD;
  if (idx >= 4 * per) {                     // zero pad row
    if (idx < 4 * per + DD) WT[idx] = (_Float16)0.f;
    return;
  }
  int m = (int)(idx / per);
  size_t r = idx % per;
  int n = (int)(r / DD), k = (int)(r % DD);
  const float* W = (m == 0) ? Wq : (m == 1) ? Wk : (m == 2) ? Wv : Wo;
  WT[(size_t)m * per + (size_t)n * DD + k] = (_Float16)W[(size_t)k * DD + n];
}

// ---------------------------------------------------------------------------
// Kernel 3: QKV GEMM (WMMA) + QK-layernorm over head dim.
// Block = 96 threads = 3 waves; wave t in {0,1,2} computes q/k/v respectively
// for the SAME 16-row x one-head tile (5 N-tiles, head padded 72->80).
// B tiles batched into distinct registers so loads pipeline ahead of WMMAs.
// ---------------------------------------------------------------------------
__global__ void __launch_bounds__(96, 1)
qkv_kernel(const _Float16* __restrict__ h,
           const _Float16* __restrict__ WT,  // WqT|WkT|WvT|WoT|zero-row
           _Float16* __restrict__ qp,        // [B][H][S][96]
           _Float16* __restrict__ kp,        // [B][H][S][96]
           _Float16* __restrict__ vt) {      // [B][H][80][S]
  const int t    = threadIdx.x >> 5;        // 0=q, 1=k, 2=v
  const int lane = threadIdx.x & 31;
  const int lh = lane & 15;
  const int hi = lane >> 4;

  int g = blockIdx.x;                        // 0 .. B*(S/16)*H - 1
  const int head = g % HH;      g /= HH;
  const int st   = g % (SS / 16);
  const int b    = g / (SS / 16);

  const _Float16* wptr = WT + (size_t)t * DD * DD;
  const _Float16* zrow = WT + 4 * (size_t)DD * DD;     // zero pad row

  vf8 acc[5];
#pragma unroll
  for (int nt = 0; nt < 5; ++nt) acc[nt] = (vf8){};

  const int arow = b * SS + st * 16 + lh;             // A: lane = row
  const _Float16* aptr = h + (size_t)arow * DD + (hi ? 16 : 0);
  const int bk = hi ? 16 : 0;                         // B: K offset by lane half

  const _Float16* brow[5];
#pragma unroll
  for (int nt = 0; nt < 5; ++nt) {
    int ncol = nt * 16 + lh;
    brow[nt] = (ncol < DH_) ? (wptr + ((size_t)(head * DH_ + ncol)) * DD + bk)
                            : (zrow + bk);
  }

  for (int kc = 0; kc < DD; kc += 32) {
    vh16 a = packA(aptr + kc);
    vh16 bm[5];
#pragma unroll
    for (int nt = 0; nt < 5; ++nt) bm[nt] = *(const vh16*)(brow[nt] + kc);
#pragma unroll
    for (int nt = 0; nt < 5; ++nt) acc[nt] = wmma16(a, bm[nt], acc[nt]);
  }

  if (t < 2) {
    float mn[8], rstd[8];
#pragma unroll
    for (int j = 0; j < 8; ++j) {
      float s = 0.f;
#pragma unroll
      for (int nt = 0; nt < 5; ++nt) s += acc[nt][j];
#pragma unroll
      for (int m = 1; m <= 8; m <<= 1) s += __shfl_xor(s, m);
      mn[j] = s * (1.f / DH_);
    }
#pragma unroll
    for (int j = 0; j < 8; ++j) {
      float v = 0.f;
#pragma unroll
      for (int nt = 0; nt < 5; ++nt) {
        bool valid = (nt < 4) || (lh < 8);            // col = nt*16+lh < 72
        float d = acc[nt][j] - mn[j];
        if (valid) v += d * d;
      }
#pragma unroll
      for (int m = 1; m <= 8; m <<= 1) v += __shfl_xor(v, m);
      rstd[j] = rsqrtf(v * (1.f / DH_) + EPSN);
    }

    _Float16* dst = (t == 0) ? qp : kp;
    const size_t base = (((size_t)(b * HH + head)) * SS + st * 16) * DHP;
#pragma unroll
    for (int nt = 0; nt < 5; ++nt) {
#pragma unroll
      for (int j = 0; j < 8; ++j) {
        int M = hi * 8 + j;                 // C layout: row = j (+8 for hi half)
        int col = nt * 16 + lh;             // C layout: col = lane
        float v = (col < DH_) ? (acc[nt][j] - mn[j]) * rstd[j] : 0.f;
        dst[base + (size_t)M * DHP + col] = (_Float16)v;
      }
    }
#pragma unroll
    for (int j = 0; j < 8; ++j) {           // zero padded cols 80..95
      int M = hi * 8 + j;
      dst[base + (size_t)M * DHP + 80 + lh] = (_Float16)0.f;
    }
  } else {
    const size_t vbase = ((size_t)(b * HH + head)) * DHV * SS;
#pragma unroll
    for (int nt = 0; nt < 5; ++nt) {
#pragma unroll
      for (int j = 0; j < 8; ++j) {
        int M = hi * 8 + j;
        int col = nt * 16 + lh;
        vt[vbase + (size_t)col * SS + st * 16 + M] = (_Float16)acc[nt][j];
      }
    }
  }
}

// ---------------------------------------------------------------------------
// Kernel 4: flash attention. 4 waves/block share (b,head); K/V chunk staged
// into LDS once per block via the Tensor Data Mover (TENSORcnt), consumed by
// all waves via ds_load_b128.  One wave = one 16-row Q tile.
// grid = B*H*(S/16) / 4 blocks, 128 threads
// ---------------------------------------------------------------------------
struct AttnShared {
  alignas(64) _Float16 kbuf[32][DHP];       // 32 keys x 96  (6144 B)
  alignas(64) _Float16 vbuf[DHV][32];       // 80 dh  x 32   (5120 B)
  alignas(64) _Float16 lp[4][16][32];       // per-wave P staging (4096 B)
};

__global__ void __launch_bounds__(128, 1)
attn_kernel(const _Float16* __restrict__ qp,
            const _Float16* __restrict__ kp,
            const _Float16* __restrict__ vt,
            _Float16* __restrict__ o) {   // [B][S][D] f16
  __shared__ AttnShared sm;

  const int tid = threadIdx.x;
  const int wid = tid >> 5;
  const int lane = tid & 31;
  const int lh = lane & 15;
  const int hi = lane >> 4;

  int g = blockIdx.x * 4 + wid;
  const int qi   = g % (SS / 16);  g /= (SS / 16);
  const int head = g % HH;
  const int b    = g / HH;

  // Q tile: 3 A chunks (K = 96), resident in registers
  vh16 aq[3];
  const size_t qrow = (((size_t)(b * HH + head)) * SS + qi * 16 + lh) * DHP;
#pragma unroll
  for (int c = 0; c < 3; ++c)
    aq[c] = packA(qp + qrow + c * 32 + (hi ? 16 : 0));

  float mrow[8], lsum[8];
  vf8 acco[5];
#pragma unroll
  for (int nt = 0; nt < 5; ++nt) acco[nt] = (vf8){};
#pragma unroll
  for (int j = 0; j < 8; ++j) { mrow[j] = -1e30f; lsum[j] = 0.f; }

  const float sc = 0.11785113019775793f;     // 1/sqrt(72)
  const _Float16* kchunk0 = kp + ((size_t)(b * HH + head)) * SS * DHP;
  const _Float16* vchunk0 = vt + ((size_t)(b * HH + head)) * DHV * SS;
  const int bk = hi ? 16 : 0;

  for (int kc2 = 0; kc2 < SS / 32; ++kc2) {  // 32-key chunks
    // ---- stage K (32x96) and V (80x32) chunk into LDS ----
#if HAVE_TDM
    if (wid == 0) {
      tdm_load_2d((unsigned)__builtin_offsetof(AttnShared, kbuf),
                  kchunk0 + (size_t)kc2 * 32 * DHP, DHP, 32, DHP);
      tdm_load_2d((unsigned)__builtin_offsetof(AttnShared, vbuf),
                  vchunk0 + kc2 * 32, 32, DHV, SS);
      tdm_wait0();
    }
    __syncthreads();
#else
    // cooperative fallback: 384 + 320 v8h moves over 128 threads
#pragma unroll
    for (int i = 0; i < 3; ++i) {
      int e = tid + i * 128;
      if (e < 384) {
        int r = e / 12, cpos = (e % 12) * 8;
        *(vh8*)&sm.kbuf[r][cpos] =
            *(const vh8*)(kchunk0 + ((size_t)(kc2 * 32 + r)) * DHP + cpos);
      }
      if (e < 320) {
        int r = e / 4, cpos = (e % 4) * 8;
        *(vh8*)&sm.vbuf[r][cpos] =
            *(const vh8*)(vchunk0 + (size_t)r * SS + kc2 * 32 + cpos);
      }
    }
    __syncthreads();
#endif

    // ---- scores: two 16x16 tiles from LDS K ----
    vf8 s0 = (vf8){}, s1 = (vf8){};
#pragma unroll
    for (int c = 0; c < 3; ++c) {
      vh16 b0 = *(const vh16*)(&sm.kbuf[lh][c * 32 + bk]);
      vh16 b1 = *(const vh16*)(&sm.kbuf[16 + lh][c * 32 + bk]);
      s0 = wmma16(aq[c], b0, s0);
      s1 = wmma16(aq[c], b1, s1);
    }
    // online softmax (row reductions across 16-lane halves)
    float ml[8], al[8], rsum[8];
#pragma unroll
    for (int j = 0; j < 8; ++j) {
      s0[j] *= sc; s1[j] *= sc;
      ml[j] = fmaxf(s0[j], s1[j]);
    }
#pragma unroll
    for (int j = 0; j < 8; ++j)
#pragma unroll
      for (int m = 1; m <= 8; m <<= 1) ml[j] = fmaxf(ml[j], __shfl_xor(ml[j], m));
#pragma unroll
    for (int j = 0; j < 8; ++j) {
      float mx = fmaxf(mrow[j], ml[j]);
      al[j] = __expf(mrow[j] - mx);
      float p0 = __expf(s0[j] - mx);
      float p1 = __expf(s1[j] - mx);
      s0[j] = p0; s1[j] = p1;
      rsum[j] = p0 + p1;
      mrow[j] = mx;
    }
#pragma unroll
    for (int j = 0; j < 8; ++j)
#pragma unroll
      for (int m = 1; m <= 8; m <<= 1) rsum[j] += __shfl_xor(rsum[j], m);
#pragma unroll
    for (int j = 0; j < 8; ++j) lsum[j] = lsum[j] * al[j] + rsum[j];
#pragma unroll
    for (int nt = 0; nt < 5; ++nt)
#pragma unroll
      for (int j = 0; j < 8; ++j) acco[nt][j] *= al[j];

    // stage P (C layout -> LDS row-major [row][key]) then reload in A layout
#pragma unroll
    for (int j = 0; j < 8; ++j) {
      int M = hi * 8 + j;
      sm.lp[wid][M][lh]      = (_Float16)s0[j];
      sm.lp[wid][M][16 + lh] = (_Float16)s1[j];
    }
    asm volatile("" ::: "memory");           // DS pipe is in-order within a wave
    vh16 pa = packA(&sm.lp[wid][lh][hi ? 16 : 0]);
    asm volatile("" ::: "memory");
#pragma unroll
    for (int nt = 0; nt < 5; ++nt) {
      vh16 vb = *(const vh16*)(&sm.vbuf[nt * 16 + lh][bk]);
      acco[nt] = wmma16(pa, vb, acco[nt]);
    }
    __syncthreads();                         // all waves done with K/V chunk
  }

  // epilogue: 1/l and scatter valid head-dim cols into o[b][s][d]
  float inv[8];
#pragma unroll
  for (int j = 0; j < 8; ++j) inv[j] = 1.f / lsum[j];
#pragma unroll
  for (int nt = 0; nt < 5; ++nt)
#pragma unroll
    for (int j = 0; j < 8; ++j) {
      int col = nt * 16 + lh;
      if (col < DH_) {
        int M = hi * 8 + j;
        o[((size_t)(b * SS + qi * 16 + M)) * DD + head * DH_ + col] =
            (_Float16)(acco[nt][j] * inv[j]);
      }
    }
}

// ---------------------------------------------------------------------------
// Kernel 5: out = alpha * (o @ Wo) + x   (WMMA GEMM + f32 epilogue)
// K-loop unrolled x2 with distinct registers so loads pipeline.
// ---------------------------------------------------------------------------
__global__ void __launch_bounds__(256, 1)
oproj_kernel(const _Float16* __restrict__ o,
             const _Float16* __restrict__ WoT,
             const float* __restrict__ x,
             const float* __restrict__ alpha,
             float* __restrict__ out) {
  const int wid = threadIdx.x >> 5;
  const int lane = threadIdx.x & 31;
  const int lh = lane & 15;
  const int hi = lane >> 4;

  int g = blockIdx.x * 8 + wid;
  const int ntile = g % (DD / 16);
  const int mtile = g / (DD / 16);           // 0 .. B*S/16 - 1
  const int b = mtile / (SS / 16);

  vf8 acc = (vf8){};
  const _Float16* aptr = o + ((size_t)(mtile * 16 + lh)) * DD + (hi ? 16 : 0);
  const _Float16* bptr = WoT + ((size_t)(ntile * 16 + lh)) * DD + (hi ? 16 : 0);

  for (int kc = 0; kc < DD; kc += 64) {
    vh16 a0 = packA(aptr + kc);
    vh16 a1 = packA(aptr + kc + 32);
    vh16 b0 = *(const vh16*)(bptr + kc);
    vh16 b1 = *(const vh16*)(bptr + kc + 32);
    acc = wmma16(a0, b0, acc);
    acc = wmma16(a1, b1, acc);
  }

  const int col = ntile * 16 + lh;
  const float av = alpha[(size_t)b * DD + col];
#pragma unroll
  for (int j = 0; j < 8; ++j) {
    int M = hi * 8 + j;
    size_t r = (size_t)(mtile * 16 + M) * DD + col;
    out[r] = av * acc[j] + x[r];
  }
}

// ---------------------------------------------------------------------------
// launch: workspace ~100.8 MB
// ---------------------------------------------------------------------------
extern "C" void kernel_launch(void* const* d_in, const int* in_sizes, int n_in,
                              void* d_out, int out_size, void* d_ws, size_t ws_size,
                              hipStream_t stream) {
  const float* x     = (const float*)d_in[0];
  const float* gamma = (const float*)d_in[1];
  const float* beta  = (const float*)d_in[2];
  const float* alpha = (const float*)d_in[3];
  const float* Wq    = (const float*)d_in[4];
  const float* Wk    = (const float*)d_in[5];
  const float* Wv    = (const float*)d_in[6];
  const float* Wo    = (const float*)d_in[7];

  char* ws = (char*)d_ws;
  const size_t hBytes  = (size_t)BB * SS * DD * 2;            // 18,874,368
  const size_t wtBytes = (4 * (size_t)DD * DD + DD) * 2;      // + zero pad row
  const size_t qkBytes = (size_t)BB * HH * SS * DHP * 2;      // 25,165,824 each

  _Float16* h  = (_Float16*)ws;                                  // also 'o'
  _Float16* WT = (_Float16*)(ws + hBytes);
  _Float16* qp = (_Float16*)(ws + hBytes + wtBytes);
  _Float16* kp = (_Float16*)(ws + hBytes + wtBytes + qkBytes);
  _Float16* vt = (_Float16*)(ws + hBytes + wtBytes + 2 * qkBytes);
  _Float16* WoT = WT + 3 * (size_t)DD * DD;

  adaln_kernel<<<BB * SS, 256, 0, stream>>>(x, gamma, beta, h);
  wtrans_kernel<<<(4 * DD * DD + DD + 255) / 256, 256, 0, stream>>>(Wq, Wk, Wv, Wo, WT);
  qkv_kernel<<<BB * (SS / 16) * HH, 96, 0, stream>>>(h, WT, qp, kp, vt);
  attn_kernel<<<(BB * HH * (SS / 16)) / 4, 128, 0, stream>>>(qp, kp, vt, h);
  oproj_kernel<<<((BB * SS / 16) * (DD / 16)) / 8, 256, 0, stream>>>(h, WoT, x, alpha,
                                                                     (float*)d_out);
}
